// NodeModel_90735479095444
// MI455X (gfx1250) — matrix operand order repaired and checked by reading, not compiled, not run on hardware
//
#include <hip/hip_runtime.h>
#include <stdint.h>

// ---------------- problem constants ----------------
#define N_NODES_C 10000
#define N_EDGES_C 640000
#define NH_C 128

// LDS row strides (in elements), padded for bank spread, 16B-aligned rows
#define IN_STRIDE 264      // ushorts, 256-wide bf16 input tile
#define N_STRIDE  136      // ushorts, 128-wide bf16 normalized tile

#define EDGE_GRID 1000     // 1000 blocks * 4 waves * 10 tiles = 40000 edge tiles

typedef __attribute__((ext_vector_type(16))) __bf16 bf16x16;
typedef __attribute__((ext_vector_type(8)))  float  f32x8;

struct BF16Frag { union { bf16x16 v; uint4 q[2]; }; };

__device__ __forceinline__ uint16_t f2bf(float f) {
    uint32_t u = __float_as_uint(f);
    u += 0x7FFFu + ((u >> 16) & 1u);   // round-to-nearest-even to bf16
    return (uint16_t)(u >> 16);
}

// wave-local LDS RAW fence: DS ops are in-order per wave; this stops compiler
// motion and waits out outstanding DS traffic before dependent loads.
__device__ __forceinline__ void lds_fence() {
    asm volatile("s_wait_dscnt 0x0" ::: "memory");
    __builtin_amdgcn_wave_barrier();
}

// ---------------- zero init ----------------
__global__ void zero_kernel(float* __restrict__ p, int n) {
    int i = blockIdx.x * 256 + threadIdx.x;
    if (i < n) p[i] = 0.0f;
}

// ---------------- weight repack: fp32 W[K][N] -> bf16 WMMA B-fragments ----------------
// Fragment f = kt*(N/16) + nt. Lane l: khi=l>>4, ncol=l&15. Element j: K=kt*32+khi*16+j.
// Stored so each lane's 16 bf16 are contiguous: out[((f*32+l)*16)+j]
__global__ void prep_kernel(const float* __restrict__ W, uint16_t* __restrict__ out,
                            int K, int N) {
    int t = blockIdx.x * 256 + threadIdx.x;
    if (t >= K * N) return;
    int j  = t & 15;
    int ln = (t >> 4) & 31;
    int f  = t >> 9;
    int ntiles = N >> 4;
    int kt = f / ntiles;
    int nt = f - kt * ntiles;
    int k = kt * 32 + (ln >> 4) * 16 + j;
    int n = nt * 16 + (ln & 15);
    out[t] = f2bf(W[(size_t)k * N + n]);
}

// A fragment (16x32 bf16) from a bf16 row-major LDS tile.
// A layout (contiguous, matches B fragments): lane<16 -> row=lane, K {b..b+15};
// lane>=16 -> row=lane-16, K {b+16..b+31}.
__device__ __forceinline__ bf16x16 load_a_frag(const uint16_t* row_base, int kbase, int khi) {
    BF16Frag a;
    const uint16_t* p = row_base + kbase + khi * 16;
    a.q[0] = *(const uint4*)(p);
    a.q[1] = *(const uint4*)(p + 8);
    return a.v;
}

__device__ __forceinline__ bf16x16 load_b_frag(const uint16_t* wfrag, int f, int lane) {
    BF16Frag b;
    const uint16_t* p = wfrag + ((size_t)(f * 32 + lane) << 4);
    b.q[0] = *(const uint4*)(p);
    b.q[1] = *(const uint4*)(p + 8);
    return b.v;
}

// In-register epilogue over WMMA D fragments:
// bias + ReLU, per-row LayerNorm via 4-stage shfl_xor butterfly (within 16-lane half),
// scale/shift, emit bf16 A-tile into LDS for the next GEMM.
__device__ __forceinline__ void epilogue_relu_ln(
        f32x8 acc[8], const float* __restrict__ bias,
        const float* __restrict__ gamma, const float* __restrict__ beta,
        uint16_t* __restrict__ sn, int khi, int mlo) {
    #pragma unroll
    for (int nt = 0; nt < 8; ++nt) {
        float b = bias[nt * 16 + mlo];
        #pragma unroll
        for (int r = 0; r < 8; ++r) {
            float v = acc[nt][r] + b;
            acc[nt][r] = v > 0.f ? v : 0.f;
        }
    }
    float s[8], ss[8];
    #pragma unroll
    for (int r = 0; r < 8; ++r) { s[r] = 0.f; ss[r] = 0.f; }
    #pragma unroll
    for (int nt = 0; nt < 8; ++nt)
        #pragma unroll
        for (int r = 0; r < 8; ++r) { float v = acc[nt][r]; s[r] += v; ss[r] += v * v; }
    #pragma unroll
    for (int off = 1; off < 16; off <<= 1)
        #pragma unroll
        for (int r = 0; r < 8; ++r) {
            s[r]  += __shfl_xor(s[r],  off, 32);
            ss[r] += __shfl_xor(ss[r], off, 32);
        }
    float mu[8], rstd[8];
    #pragma unroll
    for (int r = 0; r < 8; ++r) {
        mu[r] = s[r] * (1.0f / 128.0f);
        float var = ss[r] * (1.0f / 128.0f) - mu[r] * mu[r];
        rstd[r] = rsqrtf(var + 1e-5f);
    }
    #pragma unroll
    for (int nt = 0; nt < 8; ++nt) {
        float gg = gamma[nt * 16 + mlo];
        float bb = beta [nt * 16 + mlo];
        #pragma unroll
        for (int r = 0; r < 8; ++r) {
            float v = (acc[nt][r] - mu[r]) * rstd[r] * gg + bb;
            sn[(r + khi * 8) * N_STRIDE + nt * 16 + mlo] = f2bf(v);
        }
    }
}

// ---------------- edge MLP + scatter-add ----------------
// 4 waves/block share LDS-resident weight fragments; each wave grid-strides
// over 16-edge tiles with private staging buffers (no cross-wave barriers in loop).
__global__ __launch_bounds__(128, 2)
void edge_kernel(const float* __restrict__ x, const int* __restrict__ eidx,
                 const float* __restrict__ eattr,
                 const uint16_t* __restrict__ w1a, const float* __restrict__ b1a,
                 const float* __restrict__ g1,  const float* __restrict__ be1,
                 const uint16_t* __restrict__ w1b, const float* __restrict__ b1b,
                 float* __restrict__ sums, float* __restrict__ cnt) {
    __shared__ __align__(16) uint16_t wa_l[256 * 128];        // 64 KB W1a fragments
    __shared__ __align__(16) uint16_t wb_l[128 * 128];        // 32 KB W1b fragments
    __shared__ __align__(16) uint16_t sin_t[4][16 * IN_STRIDE];
    __shared__ __align__(16) uint16_t sn_t [4][16 * N_STRIDE];

    const int tid  = threadIdx.x;
    // ---- cooperative copy of weight fragments into LDS (once per block) ----
    for (int i = tid * 8; i < 256 * 128; i += 128 * 8)
        *(uint4*)(wa_l + i) = *(const uint4*)(w1a + i);
    for (int i = tid * 8; i < 128 * 128; i += 128 * 8)
        *(uint4*)(wb_l + i) = *(const uint4*)(w1b + i);
    __syncthreads();

    const int wave = tid >> 5;
    const int lane = tid & 31;
    const int khi  = lane >> 4;
    const int mlo  = lane & 15;
    uint16_t* sinw = sin_t[wave];
    uint16_t* snw  = sn_t[wave];
    const f32x8 vzero = {0.f, 0.f, 0.f, 0.f, 0.f, 0.f, 0.f, 0.f};

    const int nTiles = N_EDGES_C / 16;
    for (int tile = blockIdx.x * 4 + wave; tile < nTiles; tile += EDGE_GRID * 4) {
        const int e0 = tile * 16;

        // ---- stage [x[col] | edge_attr] tile as bf16 (2 lanes per edge row) ----
        {
            const int m = lane >> 1;
            const int h = lane & 1;
            const int e = e0 + m;
            const float* src = h ? (eattr + (size_t)e * NH_C)
                                 : (x + (size_t)eidx[N_EDGES_C + e] * NH_C);
            uint16_t* dst = sinw + m * IN_STRIDE + h * 128;
            #pragma unroll 8
            for (int j = 0; j < 128; j += 4) {
                float4 v = *(const float4*)(src + j);
                uint2 pk;
                pk.x = (uint32_t)f2bf(v.x) | ((uint32_t)f2bf(v.y) << 16);
                pk.y = (uint32_t)f2bf(v.z) | ((uint32_t)f2bf(v.w) << 16);
                *(uint2*)(dst + j) = pk;
            }
        }
        lds_fence();

        // ---- GEMM1: [16,256] x [256,128], B from LDS ----
        // kt stays a rolled loop: bounds live range to 1 A-frag + 8 B-frags,
        // keeping the wave under the 2-waves/SIMD VGPR budget.
        f32x8 acc[8];
        #pragma unroll
        for (int nt = 0; nt < 8; ++nt) acc[nt] = vzero;
        const uint16_t* arow = sinw + mlo * IN_STRIDE;
        #pragma unroll 1
        for (int kt = 0; kt < 8; ++kt) {
            bf16x16 a = load_a_frag(arow, kt * 32, khi);
            #pragma unroll
            for (int nt = 0; nt < 8; ++nt) {
                bf16x16 b = load_b_frag(wa_l, kt * 8 + nt, lane);
                acc[nt] = __builtin_amdgcn_wmma_f32_16x16x32_bf16(
                    false, a, false, b, (short)0, acc[nt], false, false);
            }
        }

        // ---- bias + ReLU + LayerNorm (in-register) -> bf16 LDS tile ----
        epilogue_relu_ln(acc, b1a, g1, be1, snw, khi, mlo);
        lds_fence();

        // ---- GEMM2: [16,128] x [128,128], B from LDS ----
        f32x8 acc2[8];
        #pragma unroll
        for (int nt = 0; nt < 8; ++nt) acc2[nt] = vzero;
        const uint16_t* nrow = snw + mlo * N_STRIDE;
        #pragma unroll 1
        for (int kt = 0; kt < 4; ++kt) {
            bf16x16 a = load_a_frag(nrow, kt * 32, khi);
            #pragma unroll
            for (int nt = 0; nt < 8; ++nt) {
                bf16x16 b = load_b_frag(wb_l, kt * 8 + nt, lane);
                acc2[nt] = __builtin_amdgcn_wmma_f32_16x16x32_bf16(
                    false, a, false, b, (short)0, acc2[nt], false, false);
            }
        }

        // ---- bias + scatter-add into sums[row[e]] ----
        int tgt[8];
        #pragma unroll
        for (int r = 0; r < 8; ++r) tgt[r] = eidx[e0 + r + khi * 8];
        #pragma unroll
        for (int nt = 0; nt < 8; ++nt) {
            float bias = b1b[nt * 16 + mlo];
            #pragma unroll
            for (int r = 0; r < 8; ++r)
                atomicAdd(&sums[(size_t)tgt[r] * NH_C + nt * 16 + mlo],
                          acc2[nt][r] + bias);
        }
        if (lane < 16) atomicAdd(&cnt[eidx[e0 + lane]], 1.0f);
    }
}

// ---------------- node MLP (one wave per 16-node tile) ----------------
__global__ __launch_bounds__(32, 2)
void node_kernel(const float* __restrict__ x,
                 const float* __restrict__ sums, const float* __restrict__ cnt,
                 const uint16_t* __restrict__ w2a, const float* __restrict__ b2a,
                 const float* __restrict__ g2,  const float* __restrict__ be2,
                 const uint16_t* __restrict__ w2b, const float* __restrict__ b2b,
                 float* __restrict__ out) {
    __shared__ __align__(16) uint16_t sin_t[16 * IN_STRIDE];
    __shared__ __align__(16) uint16_t sn_t [16 * N_STRIDE];

    const int lane = threadIdx.x;
    const int n0   = blockIdx.x * 16;
    const int khi  = lane >> 4;
    const int mlo  = lane & 15;

    // ---- stage [x[n] | sums[n]/max(cnt,1)] tile as bf16 ----
    {
        const int m = lane >> 1;
        const int h = lane & 1;
        const int nd = n0 + m;
        uint16_t* dst = sin_t + m * IN_STRIDE + h * 128;
        const float* src = h ? (sums + (size_t)nd * NH_C) : (x + (size_t)nd * NH_C);
        float scale = h ? (1.0f / fmaxf(cnt[nd], 1.0f)) : 1.0f;
        #pragma unroll 8
        for (int j = 0; j < 128; j += 4) {
            float4 v = *(const float4*)(src + j);
            uint2 pk;
            pk.x = (uint32_t)f2bf(v.x * scale) | ((uint32_t)f2bf(v.y * scale) << 16);
            pk.y = (uint32_t)f2bf(v.z * scale) | ((uint32_t)f2bf(v.w * scale) << 16);
            *(uint2*)(dst + j) = pk;
        }
    }
    lds_fence();

    const f32x8 vzero = {0.f, 0.f, 0.f, 0.f, 0.f, 0.f, 0.f, 0.f};

    // ---- GEMM1: [16,256] x [256,128] ----
    f32x8 acc[8];
    #pragma unroll
    for (int nt = 0; nt < 8; ++nt) acc[nt] = vzero;
    const uint16_t* arow = sin_t + mlo * IN_STRIDE;
    #pragma unroll 1
    for (int kt = 0; kt < 8; ++kt) {
        bf16x16 a = load_a_frag(arow, kt * 32, khi);
        #pragma unroll
        for (int nt = 0; nt < 8; ++nt) {
            bf16x16 b = load_b_frag(w2a, kt * 8 + nt, lane);
            acc[nt] = __builtin_amdgcn_wmma_f32_16x16x32_bf16(
                false, a, false, b, (short)0, acc[nt], false, false);
        }
    }

    epilogue_relu_ln(acc, b2a, g2, be2, sn_t, khi, mlo);
    lds_fence();

    // ---- GEMM2: [16,128] x [128,128] ----
    f32x8 acc2[8];
    #pragma unroll
    for (int nt = 0; nt < 8; ++nt) acc2[nt] = vzero;
    const uint16_t* nrow = sn_t + mlo * N_STRIDE;
    #pragma unroll 1
    for (int kt = 0; kt < 4; ++kt) {
        bf16x16 a = load_a_frag(nrow, kt * 32, khi);
        #pragma unroll
        for (int nt = 0; nt < 8; ++nt) {
            bf16x16 b = load_b_frag(w2b, kt * 8 + nt, lane);
            acc2[nt] = __builtin_amdgcn_wmma_f32_16x16x32_bf16(
                false, a, false, b, (short)0, acc2[nt], false, false);
        }
    }

    // ---- bias + store output ----
    #pragma unroll
    for (int nt = 0; nt < 8; ++nt) {
        float bias = b2b[nt * 16 + mlo];
        #pragma unroll
        for (int r = 0; r < 8; ++r)
            out[(size_t)(n0 + r + khi * 8) * NH_C + nt * 16 + mlo] = acc2[nt][r] + bias;
    }
}

// ---------------- launch ----------------
extern "C" void kernel_launch(void* const* d_in, const int* in_sizes, int n_in,
                              void* d_out, int out_size, void* d_ws, size_t ws_size,
                              hipStream_t stream) {
    const float* x     = (const float*)d_in[0];
    const int*   eidx  = (const int*)  d_in[1];
    const float* eattr = (const float*)d_in[2];
    const float* W1a = (const float*)d_in[3];
    const float* b1a = (const float*)d_in[4];
    const float* g1  = (const float*)d_in[5];
    const float* be1 = (const float*)d_in[6];
    const float* W1b = (const float*)d_in[7];
    const float* b1b = (const float*)d_in[8];
    const float* W2a = (const float*)d_in[9];
    const float* b2a = (const float*)d_in[10];
    const float* g2  = (const float*)d_in[11];
    const float* be2 = (const float*)d_in[12];
    const float* W2b = (const float*)d_in[13];
    const float* b2b = (const float*)d_in[14];
    float* out = (float*)d_out;

    // workspace layout (all 16B aligned)
    char* ws = (char*)d_ws;
    float*    sums  = (float*)   (ws);                                   // 10000*128 f32
    float*    cnt   = (float*)   (ws + (size_t)N_NODES_C * 128 * 4);     // 10000 f32
    uint16_t* w1af  = (uint16_t*)(ws + 5160000);                         // 256*128 bf16
    uint16_t* w1bf  = (uint16_t*)(ws + 5160000 + 65536);                 // 128*128 bf16
    uint16_t* w2af  = (uint16_t*)(ws + 5160000 + 65536 + 32768);         // 256*128 bf16
    uint16_t* w2bf  = (uint16_t*)(ws + 5160000 + 65536 + 32768 + 65536); // 128*128 bf16

    // zero sums + cnt (contiguous)
    {
        int n = N_NODES_C * 128 + N_NODES_C;
        zero_kernel<<<(n + 255) / 256, 256, 0, stream>>>(sums, n);
    }
    // repack weights into WMMA B-fragment order
    prep_kernel<<<(256 * 128 + 255) / 256, 256, 0, stream>>>(W1a, w1af, 256, 128);
    prep_kernel<<<(128 * 128 + 255) / 256, 256, 0, stream>>>(W1b, w1bf, 128, 128);
    prep_kernel<<<(256 * 128 + 255) / 256, 256, 0, stream>>>(W2a, w2af, 256, 128);
    prep_kernel<<<(128 * 128 + 255) / 256, 256, 0, stream>>>(W2b, w2bf, 128, 128);

    edge_kernel<<<EDGE_GRID, 128, 0, stream>>>(
        x, eidx, eattr, w1af, b1a, g1, be1, w1bf, b1b, sums, cnt);

    node_kernel<<<N_NODES_C / 16, 32, 0, stream>>>(
        x, sums, cnt, w2af, b2a, g2, be2, w2bf, b2b, out);
}
